// TimeBlocker_78348793414334
// MI455X (gfx1250) — matrix operand order repaired and checked by reading, not compiled, not run on hardware
//
#include <hip/hip_runtime.h>
#include <math.h>

// ---------------------------------------------------------------------------
// Mamba block + FC head for MI455X (gfx1250, wave32, WMMA).
// GEMMs run as split-precision bf16 ("bf16x3"): a = a_hi + a_lo, product
// a_lo*b_hi + a_hi*b_lo + a_hi*b_hi accumulated in fp32 via
// v_wmma_f32_16x16x32_bf16  -> ~16-bit mantissa, near-fp32 accuracy at
// bf16-matrix throughput. Roofline: ~450MB unique traffic => ~20us at
// 23.3 TB/s; bf16x3 compute ~45us for 128 GFLOP; fp32 WMMA would be 8x
// slower per issue -> bf16 split is the right operating point.
// k-loop has NO divergent guards (zero-padded B tiles instead), so the
// accumulators never move; fragments are two aligned 16B LDS chunks each
// (ds_load_b128), built per the ISA VGPR layouts.
// Workspace: ~139 MB float.
// ---------------------------------------------------------------------------

typedef __attribute__((ext_vector_type(16))) __bf16 v16bf;
typedef __attribute__((ext_vector_type(8)))  __bf16 v8bf;
typedef __attribute__((ext_vector_type(8)))  float  v8f;

#define D_MODEL 3072
#define D_INNER 6144
#define D_STATE 16
#define DT_RANK 192
#define BATCH   2
#define SEQ     512
#define MROWS   (BATCH * SEQ)          // 1024, multiple of 64

#define TM 64
#define TN 128
#define TK 32
#define LDP 40   // LDS row pitch in ushorts: 80B = 16B-aligned, 20 dwords
                 // -> gcd(20,64)=4, rows 0..15 hit distinct banks (no conflict)

__device__ __forceinline__ unsigned short f2bf(float f) {
    unsigned int u = __float_as_uint(f);
    u += 0x7fffu + ((u >> 16) & 1u);   // round-to-nearest-even
    return (unsigned short)(u >> 16);
}
__device__ __forceinline__ float bf2f(unsigned short s) {
    return __uint_as_float(((unsigned int)s) << 16);
}
__device__ __forceinline__ v16bf cat16(v8bf lo, v8bf hi) {
    return __builtin_shufflevector(lo, hi, 0, 1, 2, 3, 4, 5, 6, 7,
                                           8, 9, 10, 11, 12, 13, 14, 15);
}
__device__ __forceinline__ v8f wmma_bf16(v16bf a, v16bf b, v8f c) {
    // 8 args: (neg_a, A, neg_b, B, c_mod, C, reuse_a, reuse_b)
    return __builtin_amdgcn_wmma_f32_16x16x32_bf16(false, a, false, b,
                                                   (short)0, c, false, false);
}
__device__ __forceinline__ void cvt4(const float4 v, ushort4* hi, ushort4* lo) {
    ushort4 h, l;
    h.x = f2bf(v.x); l.x = f2bf(v.x - bf2f(h.x));
    h.y = f2bf(v.y); l.y = f2bf(v.y - bf2f(h.y));
    h.z = f2bf(v.z); l.z = f2bf(v.z - bf2f(h.z));
    h.w = f2bf(v.w); l.w = f2bf(v.w - bf2f(h.w));
    *hi = h; *lo = l;
}

// C[M,N] = act(A[M,K] @ W[N,K]^T + bias), ACT: 0=none 1=softplus 2=tanh
// Requires M % 64 == 0, K % 32 == 0, N % 16 == 0; 16B-aligned rows.
template <int ACT>
__global__ __launch_bounds__(256)
void gemm_bf16x3(const float* __restrict__ A, int lda,
                 const float* __restrict__ W, int ldw,
                 const float* __restrict__ bias,
                 float* __restrict__ C, int ldc,
                 int M, int N, int K)
{
    __shared__ __align__(16) unsigned short sAh[TM][LDP], sAl[TM][LDP];
    __shared__ __align__(16) unsigned short sBh[TN][LDP], sBl[TN][LDP];

    const int tid  = threadIdx.x;
    const int m0   = blockIdx.x * TM;      // M-fastest grid: W slice reuse in L2
    const int n0   = blockIdx.y * TN;
    const int wave = tid >> 5;
    const int lane = tid & 31;
    const int wm   = (wave & 3) << 4;      // 0,16,32,48
    const int wn   = (wave >> 2) << 6;     // 0,64
    const int r    = lane & 15;
    const int hh   = lane >> 4;

    v8f acc[4] = {};

    for (int kc = 0; kc < K; kc += TK) {
        // ---- stage A tile (64x32): float4 loads, hi/lo bf16 ushort4 stores ----
#pragma unroll
        for (int i = 0; i < (TM * TK / 4) / 256; ++i) {     // 2 iters
            const int e  = tid + i * 256;
            const int rr = e >> 3;                           // 8 float4 per row
            const int c4 = (e & 7) << 2;
            const float4 v = *(const float4*)&A[(size_t)(m0 + rr) * lda + kc + c4];
            ushort4 h, l; cvt4(v, &h, &l);
            *(ushort4*)&sAh[rr][c4] = h;
            *(ushort4*)&sAl[rr][c4] = l;
        }
        // ---- stage B tile (128x32), zero-padded past N (no compute guards) ----
#pragma unroll
        for (int i = 0; i < (TN * TK / 4) / 256; ++i) {     // 4 iters
            const int e  = tid + i * 256;
            const int rr = e >> 3;
            const int c4 = (e & 7) << 2;
            const int n  = n0 + rr;
            float4 v = make_float4(0.f, 0.f, 0.f, 0.f);
            if (n < N) {
                v = *(const float4*)&W[(size_t)n * ldw + kc + c4];
                if (kc + TK < K)
                    __builtin_prefetch(&W[(size_t)n * ldw + kc + TK + c4], 0, 0);
            }
            ushort4 h, l; cvt4(v, &h, &l);
            *(ushort4*)&sBh[rr][c4] = h;
            *(ushort4*)&sBl[rr][c4] = l;
        }
        __syncthreads();

        // ---- A fragment (16x32 bf16 layout: k = j + 8*hh + 8*(j>=8)):
        //      two aligned 16B chunks -> ds_load_b128 ----
        const v8bf* pah = (const v8bf*)&sAh[wm + r][0];
        const v8bf* pal = (const v8bf*)&sAl[wm + r][0];
        const v16bf ah = cat16(pah[hh], pah[hh + 2]);
        const v16bf al = cat16(pal[hh], pal[hh + 2]);
#pragma unroll
        for (int t = 0; t < 4; ++t) {
            // ---- B fragment (dense 32x16: lane=column, k = j + 8*hh + 8*(j>=8)) ----
            const v8bf* pbh = (const v8bf*)&sBh[wn + (t << 4) + r][0];
            const v8bf* pbl = (const v8bf*)&sBl[wn + (t << 4) + r][0];
            const v16bf bh = cat16(pbh[hh], pbh[hh + 2]);
            const v16bf bl = cat16(pbl[hh], pbl[hh + 2]);
            acc[t] = wmma_bf16(al, bh, acc[t]);
            acc[t] = wmma_bf16(ah, bl, acc[t]);
            acc[t] = wmma_bf16(ah, bh, acc[t]);
        }
        __syncthreads();
    }

    // ---- epilogue: C/D layout lanes 0-15 N=lane M=i, lanes 16-31 M=i+8 ----
#pragma unroll
    for (int t = 0; t < 4; ++t) {
        const int n = n0 + wn + (t << 4) + r;
        if (n < N) {
            const float bv = bias ? bias[n] : 0.0f;
#pragma unroll
            for (int i = 0; i < 8; ++i) {
                const int m = m0 + wm + i + 8 * hh;
                float v = acc[t][i] + bv;
                if (ACT == 1)      v = (v > 20.0f) ? v : log1pf(__expf(v));
                else if (ACT == 2) v = tanhf(v);
                C[(size_t)m * ldc + n] = v;
            }
        }
    }
}

// depthwise causal conv1d (K=4) + bias + SiLU on xz[..., :D_INNER] -> xi
__global__ __launch_bounds__(256)
void conv_silu_kernel(const float* __restrict__ xz,
                      const float* __restrict__ cw,
                      const float* __restrict__ cb,
                      float* __restrict__ xi)
{
    const int idx = blockIdx.x * 256 + threadIdx.x;   // < MROWS * D_INNER
    const int d  = idx % D_INNER;
    const int bl = idx / D_INNER;                      // b*SEQ + l
    const int l  = bl % SEQ;
    float acc = 0.0f;
#pragma unroll
    for (int j = 0; j < 4; ++j) {
        const int t = l - 3 + j;
        if (t >= 0)
            acc += cw[d * 4 + j] * xz[(size_t)(bl - 3 + j) * (2 * D_INNER) + d];
    }
    acc += cb[d];
    xi[(size_t)bl * D_INNER + d] = acc / (1.0f + __expf(-acc));   // SiLU
}

// selective scan: one thread per (b, d) channel; fuses +xi*D and SiLU(z) gate
__global__ __launch_bounds__(256)
void scan_kernel(const float* __restrict__ dt,
                 const float* __restrict__ xi,
                 const float* __restrict__ dbc,
                 const float* __restrict__ xz,
                 const float* __restrict__ A_log,
                 const float* __restrict__ Dp,
                 float* __restrict__ y)
{
    const int idx = blockIdx.x * 256 + threadIdx.x;    // < BATCH * D_INNER
    const int d = idx % D_INNER;
    const int b = idx / D_INNER;
    float a[D_STATE], h[D_STATE];
#pragma unroll
    for (int n = 0; n < D_STATE; ++n) {
        a[n] = -__expf(A_log[(size_t)d * D_STATE + n]);
        h[n] = 0.0f;
    }
    const float dpar = Dp[d];
    for (int l = 0; l < SEQ; ++l) {
        const size_t row = (size_t)b * SEQ + l;
        const float dtv = dt[row * D_INNER + d];
        const float xiv = xi[row * D_INNER + d];
        const float* bc = dbc + row * (DT_RANK + 2 * D_STATE);
        float acc = 0.0f;
#pragma unroll
        for (int n = 0; n < D_STATE; ++n) {
            const float dA = __expf(dtv * a[n]);
            h[n] = dA * h[n] + (dtv * bc[DT_RANK + n]) * xiv;
            acc += h[n] * bc[DT_RANK + D_STATE + n];
        }
        const float zv = xz[row * (2 * D_INNER) + D_INNER + d];
        const float g  = zv / (1.0f + __expf(-zv));
        y[row * D_INNER + d] = (acc + xiv * dpar) * g;
    }
}

extern "C" void kernel_launch(void* const* d_in, const int* in_sizes, int n_in,
                              void* d_out, int out_size, void* d_ws, size_t ws_size,
                              hipStream_t stream) {
    (void)in_sizes; (void)n_in; (void)out_size; (void)ws_size;
    const float* x      = (const float*)d_in[0];
    const float* W_in   = (const float*)d_in[1];
    const float* conv_w = (const float*)d_in[2];
    const float* conv_b = (const float*)d_in[3];
    const float* W_x    = (const float*)d_in[4];
    const float* W_dt   = (const float*)d_in[5];
    const float* b_dt   = (const float*)d_in[6];
    const float* A_log  = (const float*)d_in[7];
    const float* D_par  = (const float*)d_in[8];
    const float* W_out  = (const float*)d_in[9];
    const float* W_fc   = (const float*)d_in[10];
    const float* b_fc   = (const float*)d_in[11];
    float* out = (float*)d_out;

    float* xz  = (float*)d_ws;                               // [1024, 12288]
    float* xi  = xz  + (size_t)MROWS * (2 * D_INNER);        // [1024, 6144]
    float* dbc = xi  + (size_t)MROWS * D_INNER;              // [1024, 224]
    float* dtb = dbc + (size_t)MROWS * (DT_RANK + 2 * D_STATE); // [1024, 6144]
    float* yb  = dtb + (size_t)MROWS * D_INNER;              // [1024, 6144]
    float* y1  = yb  + (size_t)MROWS * D_INNER;              // [1024, 3072]

    const dim3 blk(256);
    const int NXZ = 2 * D_INNER, NBC = DT_RANK + 2 * D_STATE;

    // 1) xz = x @ W_in^T                          (M=1024, N=12288, K=3072)
    gemm_bf16x3<0><<<dim3(MROWS / TM, (NXZ + TN - 1) / TN), blk, 0, stream>>>(
        x, D_MODEL, W_in, D_MODEL, nullptr, xz, NXZ, MROWS, NXZ, D_MODEL);
    // 2) xi = silu(causal_conv(xz[:, :6144]) + conv_b)
    conv_silu_kernel<<<(MROWS * D_INNER) / 256, blk, 0, stream>>>(
        xz, conv_w, conv_b, xi);
    // 3) dbc = xi @ W_x^T                         (N=224, K=6144)
    gemm_bf16x3<0><<<dim3(MROWS / TM, (NBC + TN - 1) / TN), blk, 0, stream>>>(
        xi, D_INNER, W_x, D_INNER, nullptr, dbc, NBC, MROWS, NBC, D_INNER);
    // 4) dt = softplus(dbc[:, :192] @ W_dt^T + b_dt)   (N=6144, K=192)
    gemm_bf16x3<1><<<dim3(MROWS / TM, (D_INNER + TN - 1) / TN), blk, 0, stream>>>(
        dbc, NBC, W_dt, DT_RANK, b_dt, dtb, D_INNER, MROWS, D_INNER, DT_RANK);
    // 5) selective scan + skip + gate
    scan_kernel<<<(BATCH * D_INNER) / 256, blk, 0, stream>>>(
        dtb, xi, dbc, xz, A_log, D_par, yb);
    // 6) y1 = y @ W_out^T                         (N=3072, K=6144)
    gemm_bf16x3<0><<<dim3(MROWS / TM, (D_MODEL + TN - 1) / TN), blk, 0, stream>>>(
        yb, D_INNER, W_out, D_INNER, nullptr, y1, D_MODEL, MROWS, D_MODEL, D_INNER);
    // 7) out = tanh(y1 @ W_fc^T + b_fc)           (N=1024, K=3072)
    gemm_bf16x3<2><<<dim3(MROWS / TM, (1024 + TN - 1) / TN), blk, 0, stream>>>(
        y1, D_MODEL, W_fc, D_MODEL, b_fc, out, 1024, MROWS, 1024, D_MODEL);
}
